// GraphAttentionLayer_71674414235752
// MI455X (gfx1250) — matrix operand order repaired and checked by reading, not compiled, not run on hardware
//
#include <hip/hip_runtime.h>

typedef _Float16 v16h __attribute__((ext_vector_type(16)));
typedef _Float16 v8h  __attribute__((ext_vector_type(8)));
typedef float    v8f  __attribute__((ext_vector_type(8)));

#define BV    192   // B*V
#define NN    128   // neighbours
#define EE    128   // in features
#define FF    128   // out features
#define NRELT 12    // relation count (NREL+1)
#define ALPHA 0.2f

#define HSTR  136   // f16 LDS row stride (keeps 16B alignment, pads banks)
#define OSTR  133   // f32 LDS row stride (odd -> conflict-free column reads)

__global__ __launch_bounds__(128)
void gat_wmma_kernel(const float* __restrict__ feat_g,
                     const float* __restrict__ nb_g,
                     const int*   __restrict__ widx_g,
                     const float* __restrict__ ind_g,
                     const float* __restrict__ W_g,
                     const float* __restrict__ a_g,
                     float*       __restrict__ out_g)
{
    extern __shared__ char smem[];
    _Float16* s_nb   = (_Float16*)smem;                 // 128 x 136 halves (slot-sorted)
    _Float16* s_ft   = s_nb + NN * HSTR;                // 128 x 136 halves (slot-sorted)
    float*    s_nbo  = (float*)(s_ft + NN * HSTR);      // 128 x 133 f32 (original order)
    float*    s_fto  = s_nbo + NN * OSTR;               // 128 x 133 f32 (original order)
    int*      s_widx = (int*)(s_fto + NN * OSTR);       // 128
    float*    s_ind  = (float*)(s_widx + NN);           // 128
    float*    s_a    = s_ind + NN;                      // 256
    float*    s_e    = s_a + 2 * FF;                    // 128
    float*    s_attn = s_e + NN;                        // 128
    int*      s_rm   = (int*)(s_attn + NN);             // 8 per-tile relation masks
    int*      s_cnt  = s_rm + 8;                        // 12 relation counts
    int*      s_base = s_cnt + NRELT;                   // 12 slot bases
    int*      s_slot = s_base + NRELT;                  // 128: orig row -> slot
    int*      s_perm = s_slot + NN;                     // 128: slot -> orig row
    int*      s_srel = s_perm + NN;                     // 128: slot -> relation

    const int tid  = threadIdx.x;
    const int wg   = blockIdx.x;           // (b*V + v)
    const int lane = tid & 31;
    const int wave = tid >> 5;
    const int hi   = (lane >> 4) & 1;      // lane half
    const int lcol = lane & 15;

    const size_t ebase = (size_t)wg * NN * EE;
    const size_t nbase = (size_t)wg * NN;

    // ---- P0: small inputs + zero counts -----------------------------------
    const int myrel = widx_g[nbase + tid];
    s_widx[tid] = myrel;
    s_ind[tid]  = ind_g[nbase + tid];
    s_a[tid]        = a_g[tid];
    s_a[tid + 128]  = a_g[tid + 128];
    if (tid < NRELT) s_cnt[tid] = 0;
    __syncthreads();

    // ---- P1: count rows per relation (LDS atomics) ------------------------
    const int mypos = atomicAdd(&s_cnt[myrel], 1);
    __syncthreads();

    // ---- P2: exclusive prefix over 12 relations ---------------------------
    if (tid == 0) {
        int base = 0;
        for (int r = 0; r < NRELT; ++r) { s_base[r] = base; base += s_cnt[r]; }
    }
    __syncthreads();

    // ---- P3: scatter permutation ------------------------------------------
    {
        const int slot = s_base[myrel] + mypos;
        s_slot[tid]  = slot;   // orig -> slot
        s_perm[slot] = tid;    // slot -> orig
        s_srel[slot] = myrel;  // slot -> relation (sorted, runs of equal r)
    }
    __syncthreads();

    // ---- P4: per-16-slot-tile relation masks + stage embeds sorted --------
    if (tid < 8) {
        int m = 0;
        for (int i = 0; i < 16; ++i) m |= 1 << s_srel[tid * 16 + i];
        s_rm[tid] = m;
    }
    for (int it = 0; it < 32; ++it) {
        int idx4 = it * 128 + tid;             // float4 index, coalesced
        int flat = idx4 * 4;
        int row  = flat >> 7;
        int c0   = flat & 127;
        int slot = s_slot[row];
        const float4 f4 = ((const float4*)(feat_g + ebase))[idx4];
        const float4 n4 = ((const float4*)(nb_g  + ebase))[idx4];
        _Float16* pf = s_ft + slot * HSTR + c0;
        _Float16* pn = s_nb + slot * HSTR + c0;
        pf[0] = (_Float16)f4.x; pf[1] = (_Float16)f4.y;
        pf[2] = (_Float16)f4.z; pf[3] = (_Float16)f4.w;
        pn[0] = (_Float16)n4.x; pn[1] = (_Float16)n4.y;
        pn[2] = (_Float16)n4.z; pn[3] = (_Float16)n4.w;
    }
    __syncthreads();

    // ---- P5: per-relation masked GEMMs via WMMA (sorted tiles) ------------
    int relmask[8];
    int ridx[8];
#pragma unroll
    for (int tm = 0; tm < 8; ++tm) {
        relmask[tm] = s_rm[tm];
        ridx[tm]    = s_srel[tm * 16 + lcol];   // this lane's A-row relation
    }
    int relunion = 0;
#pragma unroll
    for (int tm = 0; tm < 8; ++tm) relunion |= relmask[tm];

    v16h zero16;
#pragma unroll
    for (int i = 0; i < 16; ++i) zero16[i] = (_Float16)0.0f;

    for (int tnI = 0; tnI < 2; ++tnI) {
        const int tn  = wave + tnI * 4;        // output column tile
        const int col = tn * 16 + lcol;

        v8f accN[8], accF[8];
#pragma unroll
        for (int tm = 0; tm < 8; ++tm) {
#pragma unroll
            for (int j = 0; j < 8; ++j) { accN[tm][j] = 0.0f; accF[tm][j] = 0.0f; }
        }

        for (int kb = 0; kb < EE; kb += 32) {
            for (int r = 0; r < NRELT; ++r) {
                if (!((relunion >> r) & 1)) continue;   // wave-uniform skip

                // B fragment: W[r][kb + hi*16 + h][col], h = 0..15 (L2-hot)
                v16h bfrag;
                const float* wp = W_g + ((size_t)r * EE + kb + hi * 16) * FF + col;
#pragma unroll
                for (int h = 0; h < 16; ++h) bfrag[h] = (_Float16)wp[h * FF];

#pragma unroll
                for (int tm = 0; tm < 8; ++tm) {
                    if (!((relmask[tm] >> r) & 1)) continue;  // wave-uniform;
                    // sorted rows => only ~boundary tiles have >1 relation
                    const bool keep = (ridx[tm] == r);        // per-lane row mask
                    const int rowo = (tm * 16 + lcol) * HSTR + kb + hi * 16;
                    v8h nlo = *(const v8h*)(s_nb + rowo);
                    v8h nhi = *(const v8h*)(s_nb + rowo + 8);
                    v8h flo = *(const v8h*)(s_ft + rowo);
                    v8h fhi = *(const v8h*)(s_ft + rowo + 8);
                    v16h an, af;
#pragma unroll
                    for (int i = 0; i < 8; ++i) {
                        an[i] = nlo[i]; an[i + 8] = nhi[i];
                        af[i] = flo[i]; af[i + 8] = fhi[i];
                    }
                    an = keep ? an : zero16;
                    af = keep ? af : zero16;
                    accN[tm] = __builtin_amdgcn_wmma_f32_16x16x32_f16(
                        false, an, false, bfrag, (short)0, accN[tm], false, false);
                    accF[tm] = __builtin_amdgcn_wmma_f32_16x16x32_f16(
                        false, af, false, bfrag, (short)0, accF[tm], false, false);
                }
            }
        }
        // spill C tiles to LDS, un-permuting slots back to original rows
#pragma unroll
        for (int tm = 0; tm < 8; ++tm) {
#pragma unroll
            for (int j = 0; j < 8; ++j) {
                int slot = tm * 16 + hi * 8 + j;
                int orow = s_perm[slot];
                s_nbo[orow * OSTR + col] = accN[tm][j];
                s_fto[orow * OSTR + col] = accF[tm][j];
            }
        }
    }
    __syncthreads();

    // ---- P6: attention logits + leaky ReLU --------------------------------
    {
        float eN = 0.f, eF = 0.f;
        const float* prn = s_nbo + tid * OSTR;
        const float* prf = s_fto + tid * OSTR;
        for (int f = 0; f < FF; ++f) {
            eN += prn[f] * s_a[f];
            eF += prf[f] * s_a[FF + f];
        }
        float h = eN + eF;
        s_e[tid] = (h > 0.f) ? h : ALPHA * h;
    }
    __syncthreads();

    // ---- P7: softmax over neighbours + indicator mask ---------------------
    {
        float mx = -3.4e38f;
        for (int n = 0; n < NN; ++n) mx = fmaxf(mx, s_e[n]);
        float sum = 0.f;
        for (int n = 0; n < NN; ++n) sum += expf(s_e[n] - mx);
        s_attn[tid] = expf(s_e[tid] - mx) / sum * s_ind[tid];
    }
    __syncthreads();

    // ---- P8: attention-weighted sum ---------------------------------------
    {
        float acc = 0.f;
        for (int n = 0; n < NN; ++n) acc += s_attn[n] * s_nbo[n * OSTR + tid];
        out_g[(size_t)wg * FF + tid] = acc;
    }
}

extern "C" void kernel_launch(void* const* d_in, const int* in_sizes, int n_in,
                              void* d_out, int out_size, void* d_ws, size_t ws_size,
                              hipStream_t stream) {
    (void)in_sizes; (void)n_in; (void)out_size; (void)d_ws; (void)ws_size;
    const float* feat = (const float*)d_in[0];
    const float* nb   = (const float*)d_in[1];
    const int*   widx = (const int*)d_in[2];
    const float* ind  = (const float*)d_in[3];
    const float* W    = (const float*)d_in[4];
    const float* a    = (const float*)d_in[5];
    float* out = (float*)d_out;

    const size_t lds_bytes =
        (size_t)(2 * NN * HSTR) * sizeof(_Float16) +        // s_nb, s_ft
        (size_t)(2 * NN * OSTR) * sizeof(float) +           // s_nbo, s_fto
        (size_t)NN * sizeof(int) +                          // s_widx
        (size_t)(NN + 2 * FF + NN + NN) * sizeof(float) +   // s_ind, s_a, s_e, s_attn
        (size_t)(8 + 2 * NRELT + 3 * NN) * sizeof(int);     // s_rm, s_cnt, s_base, s_slot, s_perm, s_srel

    gat_wmma_kernel<<<dim3(BV), dim3(128), lds_bytes, stream>>>(
        feat, nb, widx, ind, W, a, out);
}
